// Attention_v2_2173253451847
// MI455X (gfx1250) — matrix operand order repaired and checked by reading, not compiled, not run on hardware
//
#include <hip/hip_runtime.h>
#include <hip/hip_bf16.h>
#include <stdint.h>

// Bahdanau attention on MI455X (gfx1250, wave32, WMMA).
// Dominant GEMM (131072x1024x512, 137 GFLOP) done with v_wmma_f32_16x16x32_f16
// using a hi/lo f16 split of the f32 inputs (3 WMMAs ~ near-f32 accuracy).
// Fused epilogue: relu(+bias2) dot w_full -> e, then softmax, then alpha-weighted
// reduction of encoder_out (second 512MB stream, memory-bound).

typedef __attribute__((ext_vector_type(16))) _Float16 v16h;
typedef __attribute__((ext_vector_type(8)))  float    v8f;

#define B_SZ 32
#define N_SZ 4096
#define ENC  1024
#define DECD 512
#define ATT  512

// workspace byte offsets (needs ~2.63 MB)
#define WS_BHI   0u
#define WS_BLO   (1u << 20)
#define WS_BIAS2 (2u << 20)
#define WS_E     ((2u << 20) + (64u << 10))

__device__ __forceinline__ uint32_t pack2h(_Float16 a, _Float16 b) {
    union { _Float16 h[2]; uint32_t u; } un;
    un.h[0] = a; un.h[1] = b;   // h[0] -> bits [15:0] (even K), per ISA frag layout
    return un.u;
}

// ---------------------------------------------------------------------------
// Pack W_enc [1024 x 512] f32 into B-fragment layout, split into hi/lo f16.
// B frag (16x16x32 f16 WMMA): lane L -> N = nf*16 + (L&15); half h = L>>4;
// VGPR v holds K = 8*h + 2v + {0,1} for v<4, 16 + 8*h + 2(v-4) + {0,1} for v>=4. Linear layout:
//   d = ((nf*32 + kb)*32 + lane)*16 + (2v+p)
// so each lane's 8 VGPRs (32B) are one contiguous v16h load in the GEMM.
// ---------------------------------------------------------------------------
__global__ __launch_bounds__(256)
void pack_wenc_kernel(const float* __restrict__ W,
                      _Float16* __restrict__ Bhi, _Float16* __restrict__ Blo) {
    uint32_t d    = blockIdx.x * 256u + threadIdx.x;      // 0 .. 524287
    uint32_t e16  = d & 15u;
    uint32_t lane = (d >> 4) & 31u;
    uint32_t kb   = (d >> 9) & 31u;                       // 32 K-blocks of 32
    uint32_t nf   = d >> 14;                              // 32 N-frags of 16
    uint32_t v = e16 >> 1, p = e16 & 1u, h = lane >> 4;
    uint32_t k32  = (v < 4u ? 8u * h + 2u * v : 16u + 8u * h + 2u * (v - 4u)) + p; // B-matrix K mapping
    uint32_t K    = kb * 32u + k32;
    uint32_t Ncol = nf * 16u + (lane & 15u);
    float x = W[(size_t)K * ATT + Ncol];
    _Float16 hi = (_Float16)x;
    _Float16 lo = (_Float16)(x - (float)hi);
    Bhi[d] = hi;
    Blo[d] = lo;
}

// ---------------------------------------------------------------------------
// bias2[b,a] = decoder_hidden[b,:] @ W_dec[:,a] + b_enc[a] + b_dec[a]
// Tiny (16.8 MFLOP) -> plain VALU.
// ---------------------------------------------------------------------------
__global__ __launch_bounds__(256)
void bias2_kernel(const float* __restrict__ dec, const float* __restrict__ Wd,
                  const float* __restrict__ b_enc, const float* __restrict__ b_dec,
                  float* __restrict__ bias2) {
    uint32_t id = blockIdx.x * 256u + threadIdx.x;        // 0 .. 16383
    uint32_t a = id & (ATT - 1u), b = id >> 9;
    float acc = b_enc[a] + b_dec[a];
    const float* dr = dec + (size_t)b * DECD;
    for (int k = 0; k < DECD; ++k)
        acc += dr[k] * Wd[(size_t)k * ATT + a];
    bias2[id] = acc;
}

// ---------------------------------------------------------------------------
// Main fused kernel: per block, 64 rows x all 512 ATT cols, K=1024 in 4 chunks.
// 16 waves: wm = wave>>2 selects 16-row band, wn = wave&3 selects 128-col band.
// A staged: global f32 -> regs (pipelined) -> hi/lo f16 pairs in LDS in the
// A-fragment layout (lane L -> M = L&15; half h=L>>4; v<4: K = 8h+2v+p,
// v>=4: K = 16+8h+2(v-4)+p). 3 WMMAs per (kb,nf): AhBh + AhBl + AlBh.
// Epilogue: e[row] = sum_col relu(c + bias2)*w_full  (b_full cancels in softmax)
// ---------------------------------------------------------------------------
__global__ __launch_bounds__(512)
void att_gemm_e_kernel(const float* __restrict__ enc,
                       const _Float16* __restrict__ Bhi,
                       const _Float16* __restrict__ Blo,
                       const float* __restrict__ bias2,
                       const float* __restrict__ w_full,
                       float* __restrict__ e_out) {
    __shared__ __attribute__((aligned(32))) uint32_t sAhi[4 * 8 * 32 * 8]; // 32 KB
    __shared__ __attribute__((aligned(32))) uint32_t sAlo[4 * 8 * 32 * 8]; // 32 KB
    __shared__ float sEred[64][4];

    const uint32_t tid  = threadIdx.x;
    const uint32_t lane = tid & 31u;
    const uint32_t wv   = tid >> 5;
    const uint32_t wm   = wv >> 2;          // 0..3 : 16-row band
    const uint32_t wn   = wv & 3u;          // 0..3 : 128-col band

    const uint32_t blk   = blockIdx.x;      // 2048 blocks
    const uint32_t batch = blk >> 6;        // 64 tiles per batch
    const uint32_t tile  = blk & 63u;
    const size_t   row0  = (size_t)batch * N_SZ + (size_t)tile * 64u;

    // --- A staging assignment: thread -> (row r, K-octet j) ---
    const uint32_t r = tid >> 3;            // 0..63
    const uint32_t j = tid & 7u;            // K sub-block: k32 in [4j, 4j+4)
    const float4* Arow = (const float4*)(enc + (row0 + r) * ENC);

    // LDS-store mapping for this thread's two K pairs (k32=4j+{0,1} and +{2,3})
    uint32_t h0, v0, h1, v1;
    {
        uint32_t k32 = j * 4u;
        if (k32 < 16u) { h0 = k32 >> 3; v0 = (k32 & 7u) >> 1; }
        else { uint32_t q = k32 - 16u; h0 = q >> 3; v0 = 4u + ((q & 7u) >> 1); }
        k32 = j * 4u + 2u;
        if (k32 < 16u) { h1 = k32 >> 3; v1 = (k32 & 7u) >> 1; }
        else { uint32_t q = k32 - 16u; h1 = q >> 3; v1 = 4u + ((q & 7u) >> 1); }
    }
    const uint32_t wmr  = r >> 4;
    const uint32_t mrow = r & 15u;

    float4 st[8];
    v8f acc[8];
#pragma unroll
    for (int nf = 0; nf < 8; ++nf) acc[nf] = (v8f){0, 0, 0, 0, 0, 0, 0, 0};

    // prologue: load K-chunk 0 (64 rows x 256 K) into registers
#pragma unroll
    for (int i = 0; i < 8; ++i) st[i] = Arow[i * 8 + j];

    for (int c = 0; c < 4; ++c) {
        // convert current chunk -> hi/lo f16 pairs in fragment-layout LDS
#pragma unroll
        for (int i = 0; i < 8; ++i) {
            float4 xv = st[i];
            _Float16 ha0 = (_Float16)xv.x, hb0 = (_Float16)xv.y;
            _Float16 la0 = (_Float16)(xv.x - (float)ha0);
            _Float16 lb0 = (_Float16)(xv.y - (float)hb0);
            _Float16 ha1 = (_Float16)xv.z, hb1 = (_Float16)xv.w;
            _Float16 la1 = (_Float16)(xv.z - (float)ha1);
            _Float16 lb1 = (_Float16)(xv.w - (float)hb1);
            uint32_t base = (wmr * 8u + (uint32_t)i) * 32u;
            uint32_t idx0 = (base + h0 * 16u + mrow) * 8u + v0;
            uint32_t idx1 = (base + h1 * 16u + mrow) * 8u + v1;
            sAhi[idx0] = pack2h(ha0, hb0);
            sAlo[idx0] = pack2h(la0, lb0);
            sAhi[idx1] = pack2h(ha1, hb1);
            sAlo[idx1] = pack2h(la1, lb1);
        }
        __syncthreads();

        // issue next chunk's global loads; they overlap the WMMA loop below
        if (c < 3) {
            const float4* pn = Arow + (c + 1) * 64;
#pragma unroll
            for (int i = 0; i < 8; ++i) st[i] = pn[i * 8 + j];
        }

        // 8 k-steps x 8 n-frags x 3 WMMAs (hi*hi, hi*lo, lo*hi)
        const uint32_t kg0 = (uint32_t)c * 8u;
#pragma unroll
        for (int kb = 0; kb < 8; ++kb) {
            const uint32_t abase = ((wm * 8u + (uint32_t)kb) * 32u + lane) * 8u;
            const v16h ah = *(const v16h*)&sAhi[abase];
            const v16h al = *(const v16h*)&sAlo[abase];
#pragma unroll
            for (int nf = 0; nf < 8; ++nf) {
                size_t boff = ((size_t)(((wn * 8u + (uint32_t)nf) * 32u +
                                         (kg0 + (uint32_t)kb)) * 32u + lane)) * 16u;
                const v16h bh = *(const v16h*)(Bhi + boff);
                const v16h bl = *(const v16h*)(Blo + boff);
                acc[nf] = __builtin_amdgcn_wmma_f32_16x16x32_f16(
                    false, ah, false, bh, (short)0, acc[nf], false, false);
                acc[nf] = __builtin_amdgcn_wmma_f32_16x16x32_f16(
                    false, ah, false, bl, (short)0, acc[nf], false, false);
                acc[nf] = __builtin_amdgcn_wmma_f32_16x16x32_f16(
                    false, al, false, bh, (short)0, acc[nf], false, false);
            }
        }
        __syncthreads();
    }

    // epilogue: relu(+bias2) * w_full, reduce over 512 cols into e[64 rows]
    float sacc[8];
#pragma unroll
    for (int v = 0; v < 8; ++v) sacc[v] = 0.f;
    const uint32_t cl = lane & 15u;
#pragma unroll
    for (int nf = 0; nf < 8; ++nf) {
        uint32_t col = wn * 128u + (uint32_t)nf * 16u + cl;
        float wc = w_full[col];
        float bc = bias2[batch * ATT + col];
#pragma unroll
        for (int v = 0; v < 8; ++v) {
            float x = acc[nf][v] + bc;
            sacc[v] += fmaxf(x, 0.f) * wc;
        }
    }
#pragma unroll
    for (int v = 0; v < 8; ++v) {
        float s = sacc[v];
        s += __shfl_xor(s, 1);
        s += __shfl_xor(s, 2);
        s += __shfl_xor(s, 4);
        s += __shfl_xor(s, 8);      // reduce across the 16-lane half (wave32)
        if (cl == 0u)
            sEred[wm * 16u + (uint32_t)v + 8u * (lane >> 4)][wn] = s;
    }
    __syncthreads();
    if (tid < 64u) {
        float e = sEred[tid][0] + sEred[tid][1] + sEred[tid][2] + sEred[tid][3];
        e_out[row0 + tid] = e;      // b_full omitted: shift-invariant under softmax
    }
}

// ---------------------------------------------------------------------------
// Row softmax over N=4096 per batch -> alpha (written straight into d_out slot)
// ---------------------------------------------------------------------------
__global__ __launch_bounds__(256)
void softmax_rows_kernel(const float* __restrict__ e, float* __restrict__ alpha) {
    __shared__ float red[256];
    const uint32_t b = blockIdx.x, t = threadIdx.x;
    const float* er = e + (size_t)b * N_SZ;
    float vals[16];
    float m = -3.4e38f;
#pragma unroll
    for (int i = 0; i < 16; ++i) { vals[i] = er[t + i * 256]; m = fmaxf(m, vals[i]); }
    red[t] = m; __syncthreads();
    for (int s = 128; s > 0; s >>= 1) {
        if (t < (uint32_t)s) red[t] = fmaxf(red[t], red[t + s]);
        __syncthreads();
    }
    m = red[0]; __syncthreads();
    float sum = 0.f;
#pragma unroll
    for (int i = 0; i < 16; ++i) { vals[i] = __expf(vals[i] - m); sum += vals[i]; }
    red[t] = sum; __syncthreads();
    for (int s = 128; s > 0; s >>= 1) {
        if (t < (uint32_t)s) red[t] += red[t + s];
        __syncthreads();
    }
    float inv = 1.f / red[0];
    float* ar = alpha + (size_t)b * N_SZ;
#pragma unroll
    for (int i = 0; i < 16; ++i) ar[t + i * 256] = vals[i] * inv;
}

// ---------------------------------------------------------------------------
// awe[b,e] = sum_n alpha[b,n] * enc[b,n,e]  (memory-bound second stream)
// block = (b, 64-col band); 16 col-groups x 16 n-slices, float4 loads.
// ---------------------------------------------------------------------------
__global__ __launch_bounds__(256)
void awe_kernel(const float* __restrict__ enc, const float* __restrict__ alpha,
                float* __restrict__ awe) {
    __shared__ float red[16][64];
    const uint32_t b  = blockIdx.x >> 4;
    const uint32_t eb = blockIdx.x & 15u;
    const uint32_t t  = threadIdx.x;
    const uint32_t cg = t & 15u, ns = t >> 4;
    const uint32_t e0 = eb * 64u + cg * 4u;
    const float* ab  = alpha + (size_t)b * N_SZ;
    const float* ebp = enc + (size_t)b * N_SZ * ENC + e0;
    float4 acc = {0.f, 0.f, 0.f, 0.f};
    for (uint32_t n = ns; n < N_SZ; n += 16u) {
        float a = ab[n];
        float4 v = *(const float4*)(ebp + (size_t)n * ENC);
        acc.x += a * v.x; acc.y += a * v.y; acc.z += a * v.z; acc.w += a * v.w;
    }
    red[ns][cg * 4 + 0] = acc.x; red[ns][cg * 4 + 1] = acc.y;
    red[ns][cg * 4 + 2] = acc.z; red[ns][cg * 4 + 3] = acc.w;
    __syncthreads();
    for (int s = 8; s > 0; s >>= 1) {
        if (ns < (uint32_t)s) {
#pragma unroll
            for (int cc = 0; cc < 4; ++cc)
                red[ns][cg * 4 + cc] += red[ns + s][cg * 4 + cc];
        }
        __syncthreads();
    }
    if (t < 16u) {
#pragma unroll
        for (int cc = 0; cc < 4; ++cc)
            awe[(size_t)b * ENC + eb * 64u + t * 4u + cc] = red[0][t * 4 + cc];
    }
}

extern "C" void kernel_launch(void* const* d_in, const int* in_sizes, int n_in,
                              void* d_out, int out_size, void* d_ws, size_t ws_size,
                              hipStream_t stream) {
    (void)in_sizes; (void)n_in; (void)out_size; (void)ws_size;
    const float* enc   = (const float*)d_in[0];
    const float* dech  = (const float*)d_in[1];
    const float* Wenc  = (const float*)d_in[2];
    const float* benc  = (const float*)d_in[3];
    const float* Wdec  = (const float*)d_in[4];
    const float* bdec  = (const float*)d_in[5];
    const float* wfull = (const float*)d_in[6];
    // d_in[7] = b_full: constant shift of e, cancels in softmax -> unused.

    char* ws = (char*)d_ws;
    _Float16* Bhi  = (_Float16*)(ws + WS_BHI);
    _Float16* Blo  = (_Float16*)(ws + WS_BLO);
    float* bias2   = (float*)(ws + WS_BIAS2);
    float* e_ws    = (float*)(ws + WS_E);
    float* awe     = (float*)d_out;                 // [32,1024]
    float* alpha   = (float*)d_out + B_SZ * ENC;    // [32,4096]

    pack_wenc_kernel<<<(ENC * ATT) / 256, 256, 0, stream>>>(Wenc, Bhi, Blo);
    bias2_kernel<<<(B_SZ * ATT) / 256, 256, 0, stream>>>(dech, Wdec, benc, bdec, bias2);
    att_gemm_e_kernel<<<(B_SZ * N_SZ) / 64, 512, 0, stream>>>(enc, Bhi, Blo, bias2,
                                                              wfull, e_ws);
    softmax_rows_kernel<<<B_SZ, 256, 0, stream>>>(e_ws, alpha);
    awe_kernel<<<B_SZ * 16, 256, 0, stream>>>(enc, alpha, awe);
}
